// Phi_19971597927179
// MI455X (gfx1250) — matrix operand order repaired and checked by reading, not compiled, not run on hardware
//
#include <hip/hip_runtime.h>
#include <hip/hip_bf16.h>

typedef __attribute__((ext_vector_type(16))) _Float16 v16h;
typedef __attribute__((ext_vector_type(8)))  _Float16 v8h;
typedef __attribute__((ext_vector_type(8)))  float    v8f;

#define PP    128
#define NPIX  16384      // 128*128
#define BATCH 32
#define CINCH 8
#define CH    128        // 4*F_DIM
#define F_DIM 32

// -------------------------------------------------------------------------
// Kernel 0: pack W3 (fp32, [o][c], 128x128) into f16 WMMA A-fragment layout
// (contiguous per lane half: k = 16h + j).
// -------------------------------------------------------------------------
__global__ void pack_w3_kernel(const float* __restrict__ W3,
                               _Float16* __restrict__ w3p) {
    int t = blockIdx.x * 256 + threadIdx.x;        // 0..16383
    int j    = t & 15;
    int lane = (t >> 4) & 31;
    int kt   = (t >> 9) & 3;
    int mt   = t >> 11;                             // 0..7
    int m  = mt * 16 + (lane & 15);
    int kb = kt * 32 + ((lane & 16) ? 16 : 0);
    int K  = kb + j;                                // contiguous: k = 16h + j
    w3p[t] = (_Float16)W3[m * CH + K];
}

// -------------------------------------------------------------------------
// Kernel 1: conv1x1 (K=8) + ReLU -> r2t f16 [b][pix][c].
// One thread per pixel computes all 128 output channels and writes 256
// contiguous bytes (16x b128 stores) -> fully coalesced.
// -------------------------------------------------------------------------
__global__ void conv1_kernel(const float* __restrict__ phi,
                             const float* __restrict__ W1,
                             const float* __restrict__ b1,
                             _Float16* __restrict__ r2t) {
    __shared__ float sW1[CH * CINCH];   // 4 KB
    __shared__ float sb1[CH];

    const int tid = threadIdx.x;
    for (int i = tid; i < CH * CINCH; i += 256) sW1[i] = W1[i];
    if (tid < CH) sb1[tid] = b1[tid];
    __syncthreads();

    const int gpix = blockIdx.x * 256 + tid;        // 0..BATCH*NPIX-1
    const int b    = gpix >> 14;
    const int pix  = gpix & (NPIX - 1);

    float x[CINCH];
#pragma unroll
    for (int c = 0; c < CINCH; ++c)
        x[c] = phi[((size_t)b * CINCH + c) * NPIX + pix];

    _Float16* dst = r2t + (size_t)gpix * CH;
#pragma unroll
    for (int cg = 0; cg < CH; cg += 8) {
        v8h pack;
#pragma unroll
        for (int j = 0; j < 8; ++j) {
            const int co = cg + j;
            float acc = sb1[co];
#pragma unroll
            for (int c = 0; c < CINCH; ++c) acc += sW1[co * CINCH + c] * x[c];
            pack[j] = (_Float16)(acc > 0.f ? acc : 0.f);
        }
        *(v8h*)(dst + cg) = pack;
    }
}

// -------------------------------------------------------------------------
// Kernel 2a: per-(b,c) row sums (over w) and col sums (over h) of r2t.
// Block (b,q), 128 threads (thread = channel c): both scans are contiguous
// across threads. f32 accumulation. rowS/colS stored [b][c][q].
// -------------------------------------------------------------------------
__global__ void sums_kernel(const _Float16* __restrict__ r2t,
                            float* __restrict__ rowS,
                            float* __restrict__ colS) {
    const int b = blockIdx.x >> 7;
    const int q = blockIdx.x & 127;
    const int c = threadIdx.x;          // 0..127

    const _Float16* base = r2t + (size_t)b * NPIX * CH + c;
    float rs = 0.f, cs = 0.f;
    for (int k = 0; k < PP; ++k) {
        rs += (float)base[(size_t)(q * PP + k) * CH];   // sum over w at row h=q
        cs += (float)base[(size_t)(k * PP + q) * CH];   // sum over h at col w=q
    }
    rowS[((size_t)b * CH + c) * PP + q] = rs;
    colS[((size_t)b * CH + c) * PP + q] = cs;
}

// Kernel 2b: totals per (b,c).
__global__ void total_kernel(const float* __restrict__ rowS,
                             float* __restrict__ tot) {
    const int b = blockIdx.x;
    const int c = threadIdx.x;
    const float* r = rowS + ((size_t)b * CH + c) * PP;
    float t = 0.f;
    for (int k = 0; k < PP; ++k) t += r[k];
    tot[b * CH + c] = t;
}

// -------------------------------------------------------------------------
// Kernel 2c: exchange transform in place on r2t (channels 32..127 only;
// channels 0..31 are the identity group).  ex@x = sum(x)-x.
// One thread per pixel; contiguous 16B vector load/transform/store chunks.
// -------------------------------------------------------------------------
__global__ void transform1_kernel(_Float16* __restrict__ r2t,
                                  const float* __restrict__ rowS,
                                  const float* __restrict__ colS,
                                  const float* __restrict__ tot) {
    const int blk = blockIdx.x;
    const int b   = blk >> 6;           // 0..31
    const int pix = (blk & 63) * 256 + threadIdx.x;
    const int h = pix >> 7, w = pix & 127;

    _Float16* base = r2t + ((size_t)b * NPIX + pix) * CH;
    const float* rS = rowS + (size_t)b * CH * PP;
    const float* cS = colS + (size_t)b * CH * PP;
    const float* tS = tot  + (size_t)b * CH;

    // group 1: c in [32,64):  (colsum - v)/3
#pragma unroll
    for (int cg = 32; cg < 64; cg += 8) {
        v8h ch = *(v8h*)(base + cg);
#pragma unroll
        for (int j = 0; j < 8; ++j) {
            const int c = cg + j;
            float v = (float)ch[j];
            ch[j] = (_Float16)((cS[c * PP + w] - v) * (1.f / 3.f));
        }
        *(v8h*)(base + cg) = ch;
    }
    // group 2: c in [64,96):  (rowsum - v)/3
#pragma unroll
    for (int cg = 64; cg < 96; cg += 8) {
        v8h ch = *(v8h*)(base + cg);
#pragma unroll
        for (int j = 0; j < 8; ++j) {
            const int c = cg + j;
            float v = (float)ch[j];
            ch[j] = (_Float16)((rS[c * PP + h] - v) * (1.f / 3.f));
        }
        *(v8h*)(base + cg) = ch;
    }
    // group 3: c in [96,128): (total - rowsum - colsum + v)/9
#pragma unroll
    for (int cg = 96; cg < 128; cg += 8) {
        v8h ch = *(v8h*)(base + cg);
#pragma unroll
        for (int j = 0; j < 8; ++j) {
            const int c = cg + j;
            float v = (float)ch[j];
            ch[j] = (_Float16)((tS[c] - rS[c * PP + h] - cS[c * PP + w] + v)
                               * (1.f / 9.f));
        }
        *(v8h*)(base + cg) = ch;
    }
}

// -------------------------------------------------------------------------
// Kernel 3: conv1x1 with W3 via v_wmma_f32_16x16x32_f16 + bias + ReLU.
// Waves mt<2 produce out-channels 0..31: those are the identity group of
// postprocess-2, so they are written directly to the final output.
// -------------------------------------------------------------------------
__global__ void gemm_conv3_kernel(const _Float16* __restrict__ w3p,
                                  const _Float16* __restrict__ r2t,
                                  const float* __restrict__ b3,
                                  float* __restrict__ r3,
                                  float* __restrict__ out) {
    const int lane = threadIdx.x & 31;
    const int mt   = threadIdx.x >> 5;     // wave id == M tile (0..7)
    const int b    = blockIdx.x >> 8;      // 0..31
    const int ntg  = blockIdx.x & 255;     // group of 4 N tiles

    const _Float16* ab = w3p + ((size_t)(mt * 4) * 32 + lane) * 16;
    v16h a0 = *(const v16h*)(ab + 0 * 512);
    v16h a1 = *(const v16h*)(ab + 1 * 512);
    v16h a2 = *(const v16h*)(ab + 2 * 512);
    v16h a3 = *(const v16h*)(ab + 3 * 512);

    const int mrow = mt * 16 + ((lane & 16) ? 8 : 0);   // M of acc[r] is mrow+r
    float bias[8];
#pragma unroll
    for (int r = 0; r < 8; ++r) bias[r] = b3[mrow + r];

    const int nlo  = lane & 15;
    const int csel = (lane & 16);
    float* dstbase = (mt < 2) ? out : r3;   // identity group -> final output

#pragma unroll
    for (int s = 0; s < 4; ++s) {
        const int pix = (ntg * 4 + s) * 16 + nlo;
        const _Float16* bb = r2t + ((size_t)b * NPIX + pix) * CH + csel;

        v8f acc = {};
        v16h bv;
        bv  = *(const v16h*)(bb + 0 * 32);
        acc = __builtin_amdgcn_wmma_f32_16x16x32_f16(false, a0, false, bv,
                                                     (short)0, acc, false, false);
        bv  = *(const v16h*)(bb + 1 * 32);
        acc = __builtin_amdgcn_wmma_f32_16x16x32_f16(false, a1, false, bv,
                                                     (short)0, acc, false, false);
        bv  = *(const v16h*)(bb + 2 * 32);
        acc = __builtin_amdgcn_wmma_f32_16x16x32_f16(false, a2, false, bv,
                                                     (short)0, acc, false, false);
        bv  = *(const v16h*)(bb + 3 * 32);
        acc = __builtin_amdgcn_wmma_f32_16x16x32_f16(false, a3, false, bv,
                                                     (short)0, acc, false, false);

        float* ob = dstbase + ((size_t)b * CH + mrow) * NPIX + pix;
#pragma unroll
        for (int r = 0; r < 8; ++r) {
            float v = acc[r] + bias[r];
            ob[(size_t)r * NPIX] = v > 0.f ? v : 0.f;
        }
    }
}

// -------------------------------------------------------------------------
// Kernel 4: postprocess-2 on r3 planes (channels 32..127 only) -> out.
// -------------------------------------------------------------------------
__global__ void post2_kernel(const float* __restrict__ r3,
                             float* __restrict__ out) {
    extern __shared__ float sm[];
    float* plane  = sm;              // 16384 floats
    float* rowsum = sm + NPIX;       // 128
    float* colsum = sm + NPIX + 128; // 128
    float* totalp = sm + NPIX + 256; // 1

    const int b   = blockIdx.x / 96;
    const int co  = 32 + (blockIdx.x % 96);
    const int tid = threadIdx.x;

    const float* ib = r3 + ((size_t)b * CH + co) * NPIX;
#pragma unroll 4
    for (int i = 0; i < 64; ++i) {
        int pix = tid + i * 256;
        plane[pix] = ib[pix];
    }
    __syncthreads();

    if (tid < PP) {
        float rs = 0.f, cs = 0.f;
        for (int k = 0; k < PP; ++k) {
            rs += plane[tid * PP + k];   // rowsum[h]
            cs += plane[k * PP + tid];   // colsum[w]
        }
        rowsum[tid] = rs;
        colsum[tid] = cs;
    }
    __syncthreads();
    if (tid == 0) {
        float t = 0.f;
        for (int k = 0; k < PP; ++k) t += rowsum[k];
        *totalp = t;
    }
    __syncthreads();

    const int   g     = co >> 5;      // 1, 2 or 3
    const float total = *totalp;
    float* ob = out + ((size_t)b * CH + co) * NPIX;
#pragma unroll 4
    for (int i = 0; i < 64; ++i) {
        int pix = tid + i * 256;
        int h = pix >> 7, wq = pix & 127;
        float v = plane[pix];
        float o;
        if      (g == 1) o = (colsum[wq] - v) * (1.f / 3.f);
        else if (g == 2) o = (rowsum[h]  - v) * (1.f / 3.f);
        else             o = (total - rowsum[h] - colsum[wq] + v) * (1.f / 9.f);
        ob[pix] = o;
    }
}

// -------------------------------------------------------------------------
extern "C" void kernel_launch(void* const* d_in, const int* in_sizes, int n_in,
                              void* d_out, int out_size, void* d_ws, size_t ws_size,
                              hipStream_t stream) {
    (void)in_sizes; (void)n_in; (void)out_size; (void)ws_size;

    const float* phi = (const float*)d_in[0];   // (32, 8, 128, 128)
    const float* W1  = (const float*)d_in[1];   // (128, 8)
    const float* b1  = (const float*)d_in[2];   // (128,)
    const float* W3  = (const float*)d_in[3];   // (128, 128)
    const float* b3  = (const float*)d_in[4];   // (128,)
    // d_in[5] = ex: structure exploited analytically, not read.

    float* out = (float*)d_out;                 // (32, 128, 128, 128)

    // Workspace layout (bytes):
    char* ws = (char*)d_ws;
    size_t off = 0;
    _Float16* w3p = (_Float16*)(ws + off); off += 32768;                       // 32 KB
    _Float16* r2t = (_Float16*)(ws + off); off += (size_t)BATCH*NPIX*CH*2;     // 128 MB
    float*    r3  = (float*)(ws + off);    off += (size_t)BATCH*CH*NPIX*4;     // 256 MB
    float*    rowS= (float*)(ws + off);    off += (size_t)BATCH*CH*PP*4;       // 2 MB
    float*    colS= (float*)(ws + off);    off += (size_t)BATCH*CH*PP*4;       // 2 MB
    float*    tot = (float*)(ws + off);    off += (size_t)BATCH*CH*4;          // 16 KB

    const size_t smem = (size_t)(NPIX + 257) * sizeof(float);                  // 66564 B

    // 0) pack W3 into WMMA f16 A-fragment layout
    pack_w3_kernel<<<64, 256, 0, stream>>>(W3, w3p);

    // 1) conv1 + ReLU -> r2t (coalesced channel-contiguous stores)
    conv1_kernel<<<(BATCH * NPIX) / 256, 256, 0, stream>>>(phi, W1, b1, r2t);

    // 2) plane sums + totals + in-place exchange transform (channels 32..127)
    sums_kernel<<<BATCH * PP, PP, 0, stream>>>(r2t, rowS, colS);
    total_kernel<<<BATCH, CH, 0, stream>>>(rowS, tot);
    transform1_kernel<<<BATCH * 64, 256, 0, stream>>>(r2t, rowS, colS, tot);

    // 3) WMMA GEMM (W3) + bias + ReLU -> r3 (ch 32..127) / out (ch 0..31)
    gemm_conv3_kernel<<<BATCH * 256, 256, 0, stream>>>(w3p, r2t, b3, r3, out);

    // 4) postprocess-2 (channels 32..127) -> out
    post2_kernel<<<BATCH * 96, 256, smem, stream>>>(r3, out);
}
